// LigerFusedLinearKTOLoss_60095182405792
// MI455X (gfx1250) — matrix operand order repaired and checked by reading, not compiled, not run on hardware
//
#include <hip/hip_runtime.h>
#include <hip/hip_bf16.h>

typedef __attribute__((ext_vector_type(16))) __bf16 v16bf;
typedef __attribute__((ext_vector_type(8)))  float  v8f;

#define KTO_B      4
#define KTO_T      1024
#define KTO_R      (KTO_B * KTO_T)      // 4096 rows (tokens)
#define KTO_H      2048                 // hidden
#define KTO_V      32000                // vocab
#define KTO_NB     (KTO_V / 128)        // 250 N-blocks per row
#define KTO_IGNORE (-100)
#define KTO_BETA   0.1f

// f32 -> bf16 A fragment (16x32).  element e: K = e + (e>=8?8:0) + g*8
__device__ __forceinline__ v16bf kto_load_a(const float* __restrict__ row,
                                            int k0, int g) {
    const float* p = row + k0 + (g << 3);
    float4 q0 = *(const float4*)(p);
    float4 q1 = *(const float4*)(p + 4);
    float4 q2 = *(const float4*)(p + 16);
    float4 q3 = *(const float4*)(p + 20);
    v16bf f;
    f[0]=(__bf16)q0.x; f[1]=(__bf16)q0.y; f[2]=(__bf16)q0.z; f[3]=(__bf16)q0.w;
    f[4]=(__bf16)q1.x; f[5]=(__bf16)q1.y; f[6]=(__bf16)q1.z; f[7]=(__bf16)q1.w;
    f[8]=(__bf16)q2.x; f[9]=(__bf16)q2.y; f[10]=(__bf16)q2.z; f[11]=(__bf16)q2.w;
    f[12]=(__bf16)q3.x; f[13]=(__bf16)q3.y; f[14]=(__bf16)q3.z; f[15]=(__bf16)q3.w;
    return f;
}

// f32 -> bf16 B fragment (32x16).  element e: K = e + (e>=8?8:0) + g*8
__device__ __forceinline__ v16bf kto_load_b(const float* __restrict__ row,
                                            int k0, int g) {
    const float* p = row + k0 + (g << 3);
    float4 r0 = *(const float4*)(p);
    float4 r1 = *(const float4*)(p + 4);
    float4 r2 = *(const float4*)(p + 16);
    float4 r3 = *(const float4*)(p + 20);
    v16bf f;
    f[0]=(__bf16)r0.x; f[1]=(__bf16)r0.y; f[2]=(__bf16)r0.z; f[3]=(__bf16)r0.w;
    f[4]=(__bf16)r1.x; f[5]=(__bf16)r1.y; f[6]=(__bf16)r1.z; f[7]=(__bf16)r1.w;
    f[8]=(__bf16)r2.x; f[9]=(__bf16)r2.y; f[10]=(__bf16)r2.z; f[11]=(__bf16)r2.w;
    f[12]=(__bf16)r3.x; f[13]=(__bf16)r3.y; f[14]=(__bf16)r3.z; f[15]=(__bf16)r3.w;
    return f;
}

#define KTO_WMMA(A, Bf, C) \
    __builtin_amdgcn_wmma_f32_16x16x32_bf16(false, (A), false, (Bf), (short)0, (C), false, false)

// ---------------------------------------------------------------------------
// Fused linear + partial log-softmax statistics.
// 256 threads = 8 wave32s.  Wave w: mw = w>>2 in [0,2), nw = w&3 in [0,4).
// Each wave owns a 32x32 register tile = 2x2 of 16x16 WMMA tiles: per K-step
// it loads 2 A + 2 B bf16 fragments and issues 4 independent
// v_wmma_f32_16x16x32_bf16 (no accumulator RAW chain -> co-executes).
// Workgroup tile: M=64 rows x N=128 vocab cols.  Grid: (V/128, R/64).
// Epilogue: bias, target-logit scatter, per-row max/sum-exp over the wave's
// 32 cols, LDS merge across 4 N-waves -> one (m,s) partial per row per WG.
// ---------------------------------------------------------------------------
__global__ __launch_bounds__(256) void kto_gemm_lse(
    const float* __restrict__ X,     // (R, H)
    const float* __restrict__ W,     // (V, H)
    const float* __restrict__ bias,  // (V)
    const int*   __restrict__ tgt,   // (R)
    float* __restrict__ pm,          // (R, NB) partial row max
    float* __restrict__ ps,          // (R, NB) partial sum exp
    float* __restrict__ tl)          // (R)     target logit
{
    __shared__ float m_sh[4][64];
    __shared__ float s_sh[4][64];

    const int tid  = threadIdx.x;
    const int wave = tid >> 5;
    const int lane = tid & 31;
    const int g    = lane >> 4;        // lane-half (K-group select)
    const int hl   = lane & 15;        // lane within half
    const int mw   = wave >> 2;        // 0..1  (M sub-tile, 32 rows each)
    const int nw   = wave & 3;         // 0..3  (N sub-tile, 32 cols each)

    const int rowBase = blockIdx.y * 64 + mw * 32;
    const int colBase = blockIdx.x * 128 + nw * 32;
    const int col0 = colBase + hl;         // cols of B0/C*0
    const int col1 = colBase + 16 + hl;    // cols of B1/C*1

    const float* __restrict__ xrow0 = X + (size_t)(rowBase + hl) * KTO_H;
    const float* __restrict__ xrow1 = X + (size_t)(rowBase + 16 + hl) * KTO_H;
    const float* __restrict__ wrow0 = W + (size_t)col0 * KTO_H;
    const float* __restrict__ wrow1 = W + (size_t)col1 * KTO_H;

    v8f acc[2][2] = {{{}, {}}, {{}, {}}};

    for (int k0 = 0; k0 < KTO_H; k0 += 32) {
        v16bf a0 = kto_load_a(xrow0, k0, g);
        v16bf a1 = kto_load_a(xrow1, k0, g);
        v16bf b0 = kto_load_b(wrow0, k0, g);
        v16bf b1 = kto_load_b(wrow1, k0, g);

        // W rows are streamed once per M-block group; hint ahead.
        if (k0 + 256 < KTO_H) {
            __builtin_prefetch(wrow0 + k0 + 256, 0, 1);
            __builtin_prefetch(wrow1 + k0 + 256, 0, 1);
        }

        acc[0][0] = KTO_WMMA(a0, b0, acc[0][0]);
        acc[0][1] = KTO_WMMA(a0, b1, acc[0][1]);
        acc[1][0] = KTO_WMMA(a1, b0, acc[1][0]);
        acc[1][1] = KTO_WMMA(a1, b1, acc[1][1]);
    }

    // ---- Epilogue ----
    const float bc0 = bias[col0];
    const float bc1 = bias[col1];
#pragma unroll
    for (int ti = 0; ti < 2; ++ti) {
#pragma unroll
        for (int r = 0; r < 8; ++r) {
            // C layout: VGPR r -> row M = r + (lane>=16 ? 8 : 0), col N = lane&15.
            int row_g = rowBase + ti * 16 + r + (g << 3);
            float v0 = acc[ti][0][r] + bc0;
            float v1 = acc[ti][1][r] + bc1;
            int t = tgt[row_g];
            if (t == col0) tl[row_g] = v0;   // unique writer per valid row
            if (t == col1) tl[row_g] = v1;

            // Per-row stats over this wave's 32 columns (both N sub-tiles).
            float m = fmaxf(v0, v1);
            m = fmaxf(m, __shfl_xor(m, 8, 16));
            m = fmaxf(m, __shfl_xor(m, 4, 16));
            m = fmaxf(m, __shfl_xor(m, 2, 16));
            m = fmaxf(m, __shfl_xor(m, 1, 16));
            float e = __expf(v0 - m) + __expf(v1 - m);
            e += __shfl_xor(e, 8, 16);
            e += __shfl_xor(e, 4, 16);
            e += __shfl_xor(e, 2, 16);
            e += __shfl_xor(e, 1, 16);
            if (hl == 0) {
                int rloc = (mw << 5) + ti * 16 + r + (g << 3);
                m_sh[nw][rloc] = m;
                s_sh[nw][rloc] = e;
            }
        }
    }
    __syncthreads();

    // Merge the 4 N-waves' statistics; one (m,s) per row per workgroup.
    if (tid < 64) {
        float mm = m_sh[0][tid];
        mm = fmaxf(mm, m_sh[1][tid]);
        mm = fmaxf(mm, m_sh[2][tid]);
        mm = fmaxf(mm, m_sh[3][tid]);
        float ss = 0.0f;
#pragma unroll
        for (int w = 0; w < 4; ++w)
            ss += s_sh[w][tid] * __expf(m_sh[w][tid] - mm);
        int row_g = blockIdx.y * 64 + tid;
        size_t idx = (size_t)row_g * KTO_NB + blockIdx.x;
        pm[idx] = mm;
        ps[idx] = ss;
    }
}

// ---------------------------------------------------------------------------
// Per-row logsumexp over NB partials; emits masked per-token logp.
// One wave per row; 8 rows per 256-thread block.
// ---------------------------------------------------------------------------
__global__ __launch_bounds__(256) void kto_rowreduce(
    const float* __restrict__ pm, const float* __restrict__ ps,
    const float* __restrict__ tl, const int* __restrict__ tgt,
    float* __restrict__ lp)
{
    const int wave = threadIdx.x >> 5;
    const int lane = threadIdx.x & 31;
    const int row  = blockIdx.x * 8 + wave;

    const float* pmr = pm + (size_t)row * KTO_NB;
    const float* psr = ps + (size_t)row * KTO_NB;

    float m = -INFINITY, s = 0.0f;
    for (int i = lane; i < KTO_NB; i += 32) {
        float mi = pmr[i], si = psr[i];
        if (mi > m) { s = s * __expf(m - mi) + si; m = mi; }
        else        { s += si * __expf(mi - m); }
    }
    for (int mask = 16; mask; mask >>= 1) {
        float om = __shfl_xor(m, mask, 32);
        float os = __shfl_xor(s, mask, 32);
        if (om > m) { s = s * __expf(m - om) + os; m = om; }
        else        { s += os * __expf(om - m); }
    }
    if (lane == 0) {
        int t = tgt[row];
        float v = 0.0f;
        if (t != KTO_IGNORE) v = tl[row] - (m + __logf(s));
        lp[row] = v;
    }
}

// ---------------------------------------------------------------------------
// Final KTO combine: per-batch averages -> loss, chosen/rejected reward sums.
// Single block, deterministic tree reduction.
// ---------------------------------------------------------------------------
__global__ __launch_bounds__(256) void kto_final(
    const float* __restrict__ lp_pol, const float* __restrict__ lp_ref,
    const int* __restrict__ tgt, const int* __restrict__ labels,
    float* __restrict__ out)
{
    __shared__ float sp[256], sr[256], sc[256];
    const int tid = threadIdx.x;
    float loss = 0.0f, chosen_sum = 0.0f, rej_sum = 0.0f;

    for (int b = 0; b < KTO_B; ++b) {
        float a = 0.0f, c = 0.0f, n = 0.0f;
        for (int t = tid; t < KTO_T; t += 256) {
            int row = b * KTO_T + t;
            a += lp_pol[row];
            c += lp_ref[row];
            n += (tgt[row] != KTO_IGNORE) ? 1.0f : 0.0f;
        }
        sp[tid] = a; sr[tid] = c; sc[tid] = n;
        __syncthreads();
        for (int off = 128; off > 0; off >>= 1) {
            if (tid < off) {
                sp[tid] += sp[tid + off];
                sr[tid] += sr[tid + off];
                sc[tid] += sc[tid + off];
            }
            __syncthreads();
        }
        if (tid == 0) {
            float pol = sp[0] / sc[0];
            float ref = sr[0] / sc[0];
            float lr  = pol - ref;
            int   ch  = labels[b];
            float z   = KTO_BETA * lr;
            float arg = ch ? z : -z;
            float sig = 1.0f / (1.0f + __expf(-arg));
            loss += (1.0f - sig);
            if (ch) chosen_sum += z; else rej_sum += z;
        }
        __syncthreads();
    }
    if (tid == 0) {
        out[0] = loss / (float)KTO_B;
        out[1] = chosen_sum;
        out[2] = rej_sum;
    }
}

// ---------------------------------------------------------------------------
extern "C" void kernel_launch(void* const* d_in, const int* in_sizes, int n_in,
                              void* d_out, int out_size, void* d_ws, size_t ws_size,
                              hipStream_t stream) {
    const float* x      = (const float*)d_in[0];   // (B,T,H)
    const float* Wm     = (const float*)d_in[1];   // (V,H)
    const int*   tgt    = (const int*)  d_in[2];   // (B,T)
    const int*   labels = (const int*)  d_in[3];   // (B,)
    const float* bias   = (const float*)d_in[4];   // (V,)
    const float* rx     = (const float*)d_in[5];   // (B,T,H)
    const float* rW     = (const float*)d_in[6];   // (V,H)
    const float* rb     = (const float*)d_in[7];   // (V,)
    float* out = (float*)d_out;                    // 3 floats

    // Workspace layout (floats): 4 partial arrays (~16.4 MB) + small vectors.
    float* ws   = (float*)d_ws;
    size_t part = (size_t)KTO_R * KTO_NB;
    float* pm_p = ws;
    float* ps_p = pm_p + part;
    float* pm_r = ps_p + part;
    float* ps_r = pm_r + part;
    float* tl_p = ps_r + part;
    float* tl_r = tl_p + KTO_R;
    float* lp_p = tl_r + KTO_R;
    float* lp_r = lp_p + KTO_R;

    dim3 grid(KTO_V / 128, KTO_R / 64);
    kto_gemm_lse<<<grid, 256, 0, stream>>>(x,  Wm, bias, tgt, pm_p, ps_p, tl_p);
    kto_gemm_lse<<<grid, 256, 0, stream>>>(rx, rW, rb,   tgt, pm_r, ps_r, tl_r);

    kto_rowreduce<<<KTO_R / 8, 256, 0, stream>>>(pm_p, ps_p, tl_p, tgt, lp_p);
    kto_rowreduce<<<KTO_R / 8, 256, 0, stream>>>(pm_r, ps_r, tl_r, tgt, lp_r);

    kto_final<<<1, 256, 0, stream>>>(lp_p, lp_r, tgt, labels, out);
}
